// LSTMEncoderDecoder_42726334660942
// MI455X (gfx1250) — matrix operand order repaired and checked by reading, not compiled, not run on hardware
//
#include <hip/hip_runtime.h>
#include <hip/hip_bf16.h>
#include <math.h>

// ---------------------------------------------------------------------------
// LSTM encoder-decoder for MI455X (gfx1250), wave32 + WMMA f32_16x16x32_f16.
// B=32, T=128, E=64, H=128, V=32000.
// ---------------------------------------------------------------------------

typedef __attribute__((ext_vector_type(16))) _Float16 v16h;
typedef __attribute__((ext_vector_type(8)))  float    v8f;

constexpr int BB = 32;     // batch
constexpr int TT = 128;    // sequence length
constexpr int EE = 64;     // embed dim
constexpr int HH = 128;    // hidden dim
constexpr int VV = 32000;  // vocab
constexpr int SOS = 1;

__device__ inline v8f vzero8() {
  v8f z;
#pragma unroll
  for (int i = 0; i < 8; ++i) z[i] = 0.0f;
  return z;
}

__device__ inline v8f wmma16(v16h a, v16h b, v8f c) {
  // v_wmma_f32_16x16x32_f16: D = A(16x32) * B(32x16) + C(16x16 f32)
  return __builtin_amdgcn_wmma_f32_16x16x32_f16(
      false, a, false, b, (short)0, c, false, false);
}

// A fragment (16x32, f16) from row-major src[ld], tile origin (m0,k0).
// Contiguous k order (same as load_BT): lanes 0-15 row M=m0+lane, K=0..15; lanes 16-31 K=16..31.
__device__ inline v16h load_A(const _Float16* src, int ld, int m0, int k0, int lane) {
  int m = m0 + (lane & 15);
  int kb = k0 + ((lane >> 4) << 4);
  v16h a;
#pragma unroll
  for (int h = 0; h < 16; ++h) {
    int k = kb + h;
    a[h] = src[m * ld + k];
  }
  return a;
}

// B fragment (32x16) where B(k,n) = W[n*ldw + k] (i.e. B = W^T for row-major
// W[N,K]).  Contiguous k order as in load_A: lane holds column N=n0+(lane&15), K = base + h.
__device__ inline v16h load_BT(const _Float16* W, int ldw, int k0, int n0, int lane) {
  int n = n0 + (lane & 15);
  int kb = k0 + ((lane >> 4) << 4);
  const _Float16* p = W + n * ldw + kb;  // 16 contiguous halfs -> b128 loads
  v16h b;
#pragma unroll
  for (int h = 0; h < 16; ++h) b[h] = p[h];
  return b;
}

__device__ inline float sigm(float x) { return 1.0f / (1.0f + __expf(-x)); }

// ---------------------------------------------------------------------------
// fp32 -> f16 weight conversion (done once; fc_w_h (8.2MB) then lives in L2)
// ---------------------------------------------------------------------------
__global__ void __launch_bounds__(256) f32_to_f16_kernel(const float* __restrict__ s,
                                                         _Float16* __restrict__ d, int n) {
  int i = blockIdx.x * 256 + threadIdx.x;
  if (i < n) d[i] = (_Float16)s[i];
}

// ---------------------------------------------------------------------------
// Encoder: one persistent workgroup (8 waves). Wave w owns hidden columns
// [16w,16w+16). c-state stays in registers the whole scan; h is in LDS (f16).
// Per step per wave: 2 Mtiles x 4 gates x (2+4) ksteps = 48 WMMA.
// ---------------------------------------------------------------------------
__global__ void __launch_bounds__(256) encoder_kernel(
    const int* __restrict__ x, const float* __restrict__ embed,
    const _Float16* __restrict__ Wih, const _Float16* __restrict__ Whh,
    const float* __restrict__ bias,
    _Float16* __restrict__ h_out, float* __restrict__ c_out) {
  __shared__ _Float16 xs[BB * EE];   // 4 KB
  __shared__ _Float16 hs[BB * HH];   // 8 KB
  const int tid = threadIdx.x;
  const int lane = tid & 31;
  const int w = tid >> 5;            // wave 0..7

  for (int i = tid; i < BB * HH; i += 256) hs[i] = (_Float16)0.0f;

  const int col = w * 16 + (lane & 15);
  const float bi = bias[0 * HH + col];
  const float bf = bias[1 * HH + col];
  const float bg = bias[2 * HH + col];
  const float bo = bias[3 * HH + col];

  v8f cst[2];  // c-state fragments, batch rows per C/D layout
  cst[0] = vzero8();
  cst[1] = vzero8();
  __syncthreads();

  for (int t = 0; t < TT; ++t) {
    // gather x_t -> LDS (f16)
    for (int i = tid; i < BB * EE; i += 256) {
      int b = i / EE, e = i % EE;
      xs[i] = (_Float16)embed[(size_t)x[b * TT + t] * EE + e];
    }
    __syncthreads();

    v8f acc[2][4];
#pragma unroll
    for (int mt = 0; mt < 2; ++mt) {
#pragma unroll
      for (int g = 0; g < 4; ++g) acc[mt][g] = vzero8();
      for (int k0 = 0; k0 < EE; k0 += 32) {
        v16h a = load_A(xs, EE, mt * 16, k0, lane);
#pragma unroll
        for (int g = 0; g < 4; ++g) {
          v16h bfr = load_BT(Wih, EE, k0, g * HH + w * 16, lane);
          acc[mt][g] = wmma16(a, bfr, acc[mt][g]);
        }
      }
      for (int k0 = 0; k0 < HH; k0 += 32) {
        v16h a = load_A(hs, HH, mt * 16, k0, lane);
#pragma unroll
        for (int g = 0; g < 4; ++g) {
          v16h bfr = load_BT(Whh, HH, k0, g * HH + w * 16, lane);
          acc[mt][g] = wmma16(a, bfr, acc[mt][g]);
        }
      }
    }
    __syncthreads();  // all reads of hs for this step are done

    // LSTM nonlinearity entirely in registers; write new h into LDS
#pragma unroll
    for (int mt = 0; mt < 2; ++mt) {
#pragma unroll
      for (int r = 0; r < 8; ++r) {
        float gi = sigm(acc[mt][0][r] + bi);
        float gf = sigm(acc[mt][1][r] + bf);
        float gg = tanhf(acc[mt][2][r] + bg);
        float go = sigm(acc[mt][3][r] + bo);
        float cv = gf * cst[mt][r] + gi * gg;
        cst[mt][r] = cv;
        float hv = go * tanhf(cv);
        int m = mt * 16 + ((lane >> 4) << 3) + r;  // batch row (C/D layout)
        hs[m * HH + col] = (_Float16)hv;
      }
    }
    // next iteration's post-gather barrier makes hs writes visible
  }
  __syncthreads();

  // hand state to the decoder
#pragma unroll
  for (int mt = 0; mt < 2; ++mt)
#pragma unroll
    for (int r = 0; r < 8; ++r) {
      int m = mt * 16 + ((lane >> 4) << 3) + r;
      c_out[m * HH + col] = cst[mt][r];
    }
  for (int i = tid; i < BB * HH; i += 256) h_out[i] = hs[i];
}

// ---------------------------------------------------------------------------
// Decoder cell: one workgroup, one step. State (h f16, c f32) in workspace.
// ---------------------------------------------------------------------------
__global__ void __launch_bounds__(256) dec_cell_kernel(
    const int* __restrict__ tok, const float* __restrict__ embed,
    const _Float16* __restrict__ Wih, const _Float16* __restrict__ Whh,
    const float* __restrict__ bias,
    _Float16* __restrict__ h_bf, float* __restrict__ c_st, int t) {
  __shared__ _Float16 xs[BB * EE];
  __shared__ _Float16 hs[BB * HH];
  const int tid = threadIdx.x;
  const int lane = tid & 31;
  const int w = tid >> 5;

  // snapshot h into LDS; gather embed[token]
  for (int i = tid; i < BB * HH; i += 256) hs[i] = h_bf[i];
  for (int i = tid; i < BB * EE; i += 256) {
    int b = i / EE, e = i % EE;
    int token = (t == 0) ? SOS : tok[b];
    xs[i] = (_Float16)embed[(size_t)token * EE + e];
  }
  __syncthreads();

  const int col = w * 16 + (lane & 15);
  const float bi = bias[0 * HH + col];
  const float bf = bias[1 * HH + col];
  const float bg = bias[2 * HH + col];
  const float bo = bias[3 * HH + col];

  v8f acc[2][4];
#pragma unroll
  for (int mt = 0; mt < 2; ++mt) {
#pragma unroll
    for (int g = 0; g < 4; ++g) acc[mt][g] = vzero8();
    for (int k0 = 0; k0 < EE; k0 += 32) {
      v16h a = load_A(xs, EE, mt * 16, k0, lane);
#pragma unroll
      for (int g = 0; g < 4; ++g) {
        v16h bfr = load_BT(Wih, EE, k0, g * HH + w * 16, lane);
        acc[mt][g] = wmma16(a, bfr, acc[mt][g]);
      }
    }
    for (int k0 = 0; k0 < HH; k0 += 32) {
      v16h a = load_A(hs, HH, mt * 16, k0, lane);
#pragma unroll
      for (int g = 0; g < 4; ++g) {
        v16h bfr = load_BT(Whh, HH, k0, g * HH + w * 16, lane);
        acc[mt][g] = wmma16(a, bfr, acc[mt][g]);
      }
    }
  }

#pragma unroll
  for (int mt = 0; mt < 2; ++mt) {
#pragma unroll
    for (int r = 0; r < 8; ++r) {
      int m = mt * 16 + ((lane >> 4) << 3) + r;
      float cv = c_st[m * HH + col];
      float gi = sigm(acc[mt][0][r] + bi);
      float gf = sigm(acc[mt][1][r] + bf);
      float gg = tanhf(acc[mt][2][r] + bg);
      float go = sigm(acc[mt][3][r] + bo);
      cv = gf * cv + gi * gg;
      c_st[m * HH + col] = cv;
      h_bf[m * HH + col] = (_Float16)(go * tanhf(cv));  // hs already snapshotted
    }
  }
}

// ---------------------------------------------------------------------------
// Logits: [32,32000] = h[32,128] @ fc_w^T + fc_b.  250 blocks x 8 waves; each
// wave owns one N-tile (16 vocab cols): 2 Mtiles x 4 ksteps = 8 WMMA.
// Output write (524 MB over all t) is the HBM-bound part of this workload.
// ---------------------------------------------------------------------------
__global__ void __launch_bounds__(256) logits_kernel(
    const _Float16* __restrict__ h_bf, const _Float16* __restrict__ fcw,
    const float* __restrict__ fcb, float* __restrict__ out, int t) {
  const int lane = threadIdx.x & 31;
  const int w = threadIdx.x >> 5;
  const int n0 = (blockIdx.x * 8 + w) * 16;
  const int n = n0 + (lane & 15);
  const float bb = fcb[n];

#pragma unroll
  for (int mt = 0; mt < 2; ++mt) {
    v8f acc = vzero8();
    for (int k0 = 0; k0 < HH; k0 += 32) {
      v16h a = load_A(h_bf, HH, mt * 16, k0, lane);
      v16h bfr = load_BT(fcw, HH, k0, n0, lane);
      acc = wmma16(a, bfr, acc);
    }
#pragma unroll
    for (int r = 0; r < 8; ++r) {
      int b = mt * 16 + ((lane >> 4) << 3) + r;
      out[((size_t)b * TT + t) * VV + n] = acc[r] + bb;
    }
  }
}

// ---------------------------------------------------------------------------
// Greedy argmax over the vocab (first-max tie-break like jnp.argmax).
// ---------------------------------------------------------------------------
__global__ void __launch_bounds__(256) argmax_kernel(const float* __restrict__ out,
                                                     int* __restrict__ tok, int t) {
  const int b = blockIdx.x;
  const int tid = threadIdx.x;
  const float* row = out + ((size_t)b * TT + t) * VV;

  float best = -INFINITY;
  int bidx = VV;
  for (int i = tid; i < VV; i += 256) {
    float v = row[i];
    if (v > best || (v == best && i < bidx)) { best = v; bidx = i; }
  }
  __shared__ float sv[256];
  __shared__ int si[256];
  sv[tid] = best;
  si[tid] = bidx;
  __syncthreads();
  for (int s = 128; s > 0; s >>= 1) {
    if (tid < s) {
      if (sv[tid + s] > sv[tid] ||
          (sv[tid + s] == sv[tid] && si[tid + s] < si[tid])) {
        sv[tid] = sv[tid + s];
        si[tid] = si[tid + s];
      }
    }
    __syncthreads();
  }
  if (tid == 0) tok[b] = si[0];
}

// ---------------------------------------------------------------------------
extern "C" void kernel_launch(void* const* d_in, const int* in_sizes, int n_in,
                              void* d_out, int out_size, void* d_ws, size_t ws_size,
                              hipStream_t stream) {
  (void)in_sizes; (void)n_in; (void)out_size; (void)ws_size;
  const int*   x       = (const int*)d_in[0];    // [32,128]
  const float* embed   = (const float*)d_in[1];  // [32000,64]
  const float* enc_Wih = (const float*)d_in[2];  // [512,64]
  const float* enc_Whh = (const float*)d_in[3];  // [512,128]
  const float* enc_b   = (const float*)d_in[4];  // [512]
  const float* dec_Wih = (const float*)d_in[5];
  const float* dec_Whh = (const float*)d_in[6];
  const float* dec_b   = (const float*)d_in[7];
  const float* fc_w    = (const float*)d_in[8];  // [32000,128]
  const float* fc_b    = (const float*)d_in[9];  // [32000]
  float* out = (float*)d_out;                    // [32,128,32000]

  // workspace carve-up (256B aligned); total ~8.6 MB
  size_t off = 0;
  char* base = (char*)d_ws;
  auto carve = [&](size_t bytes) {
    void* p = base + off;
    off += (bytes + 255) & ~(size_t)255;
    return p;
  };
  _Float16* encWih_h = (_Float16*)carve(512 * 64 * 2);
  _Float16* encWhh_h = (_Float16*)carve(512 * 128 * 2);
  _Float16* decWih_h = (_Float16*)carve(512 * 64 * 2);
  _Float16* decWhh_h = (_Float16*)carve(512 * 128 * 2);
  _Float16* fcw_h    = (_Float16*)carve((size_t)VV * HH * 2);
  _Float16* h_bf     = (_Float16*)carve(BB * HH * 2);
  float*    c_st     = (float*)carve(BB * HH * 4);
  int*      tok      = (int*)carve(BB * 4);

  auto cv = [&](const float* s, _Float16* d, int n) {
    f32_to_f16_kernel<<<(n + 255) / 256, 256, 0, stream>>>(s, d, n);
  };
  cv(enc_Wih, encWih_h, 512 * 64);
  cv(enc_Whh, encWhh_h, 512 * 128);
  cv(dec_Wih, decWih_h, 512 * 64);
  cv(dec_Whh, decWhh_h, 512 * 128);
  cv(fc_w,    fcw_h,    VV * HH);

  encoder_kernel<<<1, 256, 0, stream>>>(x, embed, encWih_h, encWhh_h, enc_b,
                                        h_bf, c_st);

  for (int t = 0; t < TT; ++t) {
    dec_cell_kernel<<<1, 256, 0, stream>>>(tok, embed, decWih_h, decWhh_h,
                                           dec_b, h_bf, c_st, t);
    logits_kernel<<<VV / 16 / 8, 256, 0, stream>>>(h_bf, fcw_h, fc_b, out, t);
    argmax_kernel<<<BB, 256, 0, stream>>>(out, tok, t);
  }
}
